// TimeGraphConv_20856361189430
// MI455X (gfx1250) — matrix operand order repaired and checked by reading, not compiled, not run on hardware
//
#include <hip/hip_runtime.h>

typedef __bf16 bf16_t;
typedef __attribute__((ext_vector_type(16))) __bf16 bf16x16;
typedef __attribute__((ext_vector_type(8)))  __bf16 bf16x8;
typedef __attribute__((ext_vector_type(4)))  __bf16 bf16x4;
typedef __attribute__((ext_vector_type(8)))  float  f32x8;
typedef __attribute__((ext_vector_type(4)))  float  f32x4;
typedef __attribute__((ext_vector_type(4)))  int    v4i;

// address-space qualified int4 pointees for the async-copy builtin
typedef v4i __attribute__((address_space(1))) v4i_g;
typedef v4i __attribute__((address_space(3))) v4i_l;

#define NNODE 64
#define BB    128
#define LL    4096
#define EE    16

__device__ __forceinline__ bf16_t to_bf16(float f) {
    union { float f; unsigned u; } v; v.f = f;
    unsigned r = (v.u + 0x7FFFu + ((v.u >> 16) & 1u)) >> 16;
    unsigned short s = (unsigned short)r;
    return __builtin_bit_cast(bf16_t, s);
}

// ---- CDNA5 async global->LDS copy (16B per lane), ASYNCcnt-tracked ----
__device__ __forceinline__ void async_cp16(bf16_t* dst_lds, const bf16_t* src_glb) {
#if defined(__gfx1250__) && __has_builtin(__builtin_amdgcn_global_load_async_to_lds_b128)
    __builtin_amdgcn_global_load_async_to_lds_b128(
        (v4i_g*)(const_cast<bf16_t*>(src_glb)),
        (v4i_l*)(dst_lds),
        0, 0);
#else
    *(bf16x8*)dst_lds = *(const bf16x8*)src_glb;
#endif
}

__device__ __forceinline__ void wait_async0() {
#if defined(__gfx1250__) && __has_builtin(__builtin_amdgcn_s_wait_asynccnt)
    __builtin_amdgcn_s_wait_asynccnt(0);
#elif defined(__gfx1250__) && __has_builtin(__builtin_amdgcn_global_load_async_to_lds_b128)
    asm volatile("s_wait_asynccnt 0" ::: "memory");
#endif
}

// ---------------- fp32 -> bf16 conversion (4-wide, NT reads) ----------------
__global__ __launch_bounds__(256) void k_cvt(const float* __restrict__ s,
                                             bf16_t* __restrict__ d, int n4) {
    int i = blockIdx.x * 256 + threadIdx.x;
    if (i < n4) {
        f32x4 v = __builtin_nontemporal_load((const f32x4*)s + i);
        bf16x4 o;
        o.x = to_bf16(v.x); o.y = to_bf16(v.y);
        o.z = to_bf16(v.z); o.w = to_bf16(v.w);
        *(bf16x4*)(d + (size_t)i * 4) = o;
    }
}

// ---- Wl/Wr (L,E) fp32 -> (E,L) bf16 transpose-convert ----
__global__ __launch_bounds__(256) void k_cvtT(const float* __restrict__ W,
                                              bf16_t* __restrict__ WT) {
    int i = blockIdx.x * 256 + threadIdx.x;   // i = e*L + l
    int l = i & (LL - 1);
    int e = i >> 12;
    WT[i] = to_bf16(W[l * EE + e]);
}

// ---------------- att = feat @ W + b : M=8192, N=16, K=4096 ----------------
// one wave per 16-row M tile; direct-from-global operand builds
__global__ __launch_bounds__(256) void k_att(const bf16_t* __restrict__ featB,
                                             const bf16_t* __restrict__ WT,
                                             const float* __restrict__ bvec,
                                             float* __restrict__ att) {
    const int lane = threadIdx.x & 31;
    const int wave = threadIdx.x >> 5;
    const int m0   = (blockIdx.x * 8 + wave) * 16;
    const int half = lane >> 4;   // 0 | 1
    const int lid  = lane & 15;
    f32x8 acc = {};
    const bf16_t* arow = featB + (size_t)(m0 + lid) * LL;
    const bf16_t* brow = WT + (size_t)lid * LL;
    for (int k0 = 0; k0 < LL; k0 += 32) {
        union { bf16x8 p[2]; bf16x16 v; } A, Bm;
        A.p[0]  = *(const bf16x8*)(arow + k0 + half * 8);
        A.p[1]  = *(const bf16x8*)(arow + k0 + 16 + half * 8);
        Bm.p[0] = *(const bf16x8*)(brow + k0 + half * 8);
        Bm.p[1] = *(const bf16x8*)(brow + k0 + 16 + half * 8);
        acc = __builtin_amdgcn_wmma_f32_16x16x32_bf16(false, A.v, false, Bm.v,
                                                      (short)0, acc, false, false);
    }
    float bv = bvec[lid];
    for (int j = 0; j < 8; ++j) {
        int m = m0 + j + half * 8;
        att[(size_t)m * EE + lid] = acc[j] + bv;
    }
}

// ---------------- out = relu(feat @ Wm^T + bm) : 8192 x 4096 x K4096 ----------------
// double-buffered: async global->LDS staging of K-step k+1 overlaps WMMA on k
#define LDA 40   // 32 + 8 halves pad (80B rows, 16B aligned)
#define LDB 40
__global__ __launch_bounds__(256) void k_main(const bf16_t* __restrict__ featB,
                                              const bf16_t* __restrict__ WmB,
                                              const float* __restrict__ bm,
                                              bf16_t* __restrict__ outB) {
    __shared__ bf16_t lA[2][128 * LDA];
    __shared__ bf16_t lB[2][256 * LDB];
    const int tid  = threadIdx.x;
    const int lane = tid & 31, wave = tid >> 5;
    const int mBase = blockIdx.x * 128;
    const int nBase = blockIdx.y * 256;
    const int wm = wave >> 2, wn = wave & 3;     // 2 x 4 wave grid
    const int m0w = wm * 64, n0w = wn * 64;
    const int half = lane >> 4, lid = lane & 15;

    f32x8 acc[4][4] = {};

    auto stage = [&](int buf, int kb) {
        for (int i = 0; i < 2; ++i) {
            int c = tid + i * 256;               // 512 chunks of 16B
            int row = c >> 2, part = c & 3;
            async_cp16(&lA[buf][row * LDA + part * 8],
                       featB + (size_t)(mBase + row) * LL + kb + part * 8);
        }
        for (int i = 0; i < 4; ++i) {
            int c = tid + i * 256;               // 1024 chunks of 16B
            int row = c >> 2, part = c & 3;
            async_cp16(&lB[buf][row * LDB + part * 8],
                       WmB + (size_t)(nBase + row) * LL + kb + part * 8);
        }
    };

    stage(0, 0);
    wait_async0();
    __syncthreads();

    for (int kb = 0; kb < LL; kb += 32) {
        const int cur = (kb >> 5) & 1;
        if (kb + 32 < LL) stage(cur ^ 1, kb + 32);

        union { bf16x8 p[2]; bf16x16 v; } Af[4], Bf[4];
        for (int t = 0; t < 4; ++t) {
            int m = m0w + t * 16 + lid;
            Af[t].p[0] = *(const bf16x8*)&lA[cur][m * LDA + half * 8];
            Af[t].p[1] = *(const bf16x8*)&lA[cur][m * LDA + 16 + half * 8];
            int n = n0w + t * 16 + lid;
            Bf[t].p[0] = *(const bf16x8*)&lB[cur][n * LDB + half * 8];
            Bf[t].p[1] = *(const bf16x8*)&lB[cur][n * LDB + 16 + half * 8];
        }
        for (int mt = 0; mt < 4; ++mt)
            for (int nt = 0; nt < 4; ++nt)
                acc[mt][nt] = __builtin_amdgcn_wmma_f32_16x16x32_bf16(
                    false, Af[mt].v, false, Bf[nt].v, (short)0, acc[mt][nt], false, false);

        wait_async0();
        __syncthreads();
    }

    for (int mt = 0; mt < 4; ++mt)
        for (int nt = 0; nt < 4; ++nt) {
            int n = nBase + n0w + nt * 16 + lid;
            float bmv = bm[n];
            for (int j = 0; j < 8; ++j) {
                int m = mBase + m0w + mt * 16 + j + half * 8;
                float v = acc[mt][nt][j] + bmv;
                v = fmaxf(v, 0.0f);
                outB[(size_t)m * LL + n] = to_bf16(v);
            }
        }
}

// -------- e[d,s,b] = att_l[s,b]·att_r[d,b], then BN over b (channel = d*64+s) --------
__global__ __launch_bounds__(128) void k_edge(const float* __restrict__ attL,
                                              const float* __restrict__ attR,
                                              const float* __restrict__ gamma,
                                              const float* __restrict__ beta,
                                              float* __restrict__ ebn) {
    const int ch = blockIdx.x;            // 4096 = d*64+s
    const int d = ch >> 6, s = ch & 63;
    const int b = threadIdx.x;            // 128
    const float* al = attL + ((size_t)s * BB + b) * EE;
    const float* ar = attR + ((size_t)d * BB + b) * EE;
    float x = 0.0f;
    for (int e = 0; e < EE; ++e) x += al[e] * ar[e];
    __shared__ float red[128];
    red[b] = x; __syncthreads();
    for (int off = 64; off > 0; off >>= 1) {
        if (b < off) red[b] += red[b + off];
        __syncthreads();
    }
    float mu = red[0] * (1.0f / 128.0f);
    __syncthreads();
    float dx = x - mu;
    red[b] = dx * dx; __syncthreads();
    for (int off = 64; off > 0; off >>= 1) {
        if (b < off) red[b] += red[b + off];
        __syncthreads();
    }
    float var = red[0] * (1.0f / 128.0f);
    float y = gamma[ch] * dx * rsqrtf(var + 1e-5f) + beta[ch];
    ebn[(size_t)ch * BB + b] = y;
}

// -------- softmax over s per (d,b); emit bf16 w in (b,d,s) layout --------
__global__ __launch_bounds__(128) void k_soft(const float* __restrict__ ebn,
                                              bf16_t* __restrict__ wB) {
    const int d = blockIdx.x;             // 64
    const int b = threadIdx.x;            // 128
    const float* col = ebn + (size_t)d * 64 * BB + b;   // stride BB over s
    float mx = -3.0e38f;
    for (int s = 0; s < 64; ++s) mx = fmaxf(mx, col[(size_t)s * BB]);
    float sum = 0.0f;
    for (int s = 0; s < 64; ++s) sum += __expf(col[(size_t)s * BB] - mx);
    float inv = 1.0f / sum;
    bf16_t* o = wB + ((size_t)b * 64 + d) * 64;
    for (int s = 0; s < 64; ++s) o[s] = to_bf16(__expf(col[(size_t)s * BB] - mx) * inv);
}

// -------- rst[d,b,l] = sum_s w[d,s,b] * out[s,b,l] + bias[l] : per-b 64x4096 K=64 --------
#define LDV 72   // 64 + 8 halves pad (144B rows, 16B aligned)
__global__ __launch_bounds__(256) void k_agg(const bf16_t* __restrict__ wB,
                                             const bf16_t* __restrict__ outB,
                                             const float* __restrict__ bias,
                                             float* __restrict__ rst) {
    __shared__ bf16_t lA[64 * LDV];       // (d, s) k-contiguous
    __shared__ bf16_t lB[256 * LDV];      // (l_local, s) k-contiguous
    const int b  = blockIdx.x;            // 128
    const int l0 = blockIdx.y * 256;      // 16
    const int tid = threadIdx.x, lane = tid & 31, wave = tid >> 5;
    const int n0w = wave * 32;            // 8 waves cover N=256
    const int half = lane >> 4, lid = lane & 15;

    const bf16_t* wsrc = wB + (size_t)b * 64 * 64;
    for (int i = 0; i < 2; ++i) {         // 512 x 16B chunks
        int c = tid + i * 256;
        int row = c >> 3, part = c & 7;
        *(bf16x8*)&lA[row * LDV + part * 8] = *(const bf16x8*)(wsrc + row * 64 + part * 8);
    }
    for (int i = 0; i < 8; ++i) {         // 2048 chunks; transpose (s,l) -> (l,s)
        int c = tid + i * 256;
        int s = c >> 5, part = c & 31;
        bf16x8 v = *(const bf16x8*)(outB + ((size_t)s * BB + b) * LL + l0 + part * 8);
        int lb = part * 8;
        for (int q = 0; q < 8; ++q) lB[(lb + q) * LDV + s] = v[q];
    }
    __syncthreads();

    f32x8 acc[4][2] = {};
    for (int ks = 0; ks < 2; ++ks) {
        union { bf16x8 p[2]; bf16x16 v; } Af[4], Bf[2];
        for (int mt = 0; mt < 4; ++mt) {
            int m = mt * 16 + lid;
            Af[mt].p[0] = *(const bf16x8*)&lA[m * LDV + ks * 32 + half * 8];
            Af[mt].p[1] = *(const bf16x8*)&lA[m * LDV + ks * 32 + 16 + half * 8];
        }
        for (int nt = 0; nt < 2; ++nt) {
            int n = n0w + nt * 16 + lid;
            Bf[nt].p[0] = *(const bf16x8*)&lB[n * LDV + ks * 32 + half * 8];
            Bf[nt].p[1] = *(const bf16x8*)&lB[n * LDV + ks * 32 + 16 + half * 8];
        }
        for (int mt = 0; mt < 4; ++mt)
            for (int nt = 0; nt < 2; ++nt)
                acc[mt][nt] = __builtin_amdgcn_wmma_f32_16x16x32_bf16(
                    false, Af[mt].v, false, Bf[nt].v, (short)0, acc[mt][nt], false, false);
    }
    for (int mt = 0; mt < 4; ++mt)
        for (int nt = 0; nt < 2; ++nt) {
            int l = l0 + n0w + nt * 16 + lid;
            float bv = bias[l];
            for (int j = 0; j < 8; ++j) {
                int dd = mt * 16 + j + half * 8;
                __builtin_nontemporal_store(acc[mt][nt][j] + bv,
                                            rst + ((size_t)dd * BB + b) * LL + l);
            }
        }
}

extern "C" void kernel_launch(void* const* d_in, const int* in_sizes, int n_in,
                              void* d_out, int out_size, void* d_ws, size_t ws_size,
                              hipStream_t stream) {
    const float* feat  = (const float*)d_in[0];
    const float* Wl    = (const float*)d_in[1];
    const float* bl    = (const float*)d_in[2];
    const float* Wr    = (const float*)d_in[3];
    const float* br    = (const float*)d_in[4];
    const float* Wm    = (const float*)d_in[5];
    const float* bm    = (const float*)d_in[6];
    const float* gamma = (const float*)d_in[7];
    const float* beta  = (const float*)d_in[8];
    const float* bias  = (const float*)d_in[9];

    char* ws = (char*)d_ws;
    // workspace layout (bytes)
    const size_t FEATB_N = (size_t)NNODE * BB * LL;        // 33554432
    const size_t WM_N    = (size_t)LL * LL;                // 16777216
    bf16_t* featB = (bf16_t*)(ws);                                    size_t off = FEATB_N * 2;
    bf16_t* WmB   = (bf16_t*)(ws + off);                              off += WM_N * 2;
    bf16_t* WlT   = (bf16_t*)(ws + off);                              off += (size_t)EE * LL * 2;
    bf16_t* WrT   = (bf16_t*)(ws + off);                              off += (size_t)EE * LL * 2;
    float*  attL  = (float*)(ws + off);                               off += (size_t)NNODE * BB * EE * 4;
    float*  attR  = (float*)(ws + off);                               off += (size_t)NNODE * BB * EE * 4;
    float*  ebn   = (float*)(ws + off);                               off += (size_t)NNODE * NNODE * BB * 4;
    bf16_t* wBuf  = (bf16_t*)(ws + off);                              off += (size_t)BB * NNODE * NNODE * 2;
    bf16_t* outB  = (bf16_t*)(ws + off);                              off += FEATB_N * 2;

    // conversions
    k_cvt <<<(unsigned)(FEATB_N / 4 / 256), 256, 0, stream>>>(feat, featB, (int)(FEATB_N / 4));
    k_cvt <<<(unsigned)(WM_N / 4 / 256),    256, 0, stream>>>(Wm,   WmB,   (int)(WM_N / 4));
    k_cvtT<<<(EE * LL) / 256, 256, 0, stream>>>(Wl, WlT);
    k_cvtT<<<(EE * LL) / 256, 256, 0, stream>>>(Wr, WrT);

    // attention projections (WMMA, N = E = 16)
    k_att<<<NNODE * BB / 16 / 8, 256, 0, stream>>>(featB, WlT, bl, attL);
    k_att<<<NNODE * BB / 16 / 8, 256, 0, stream>>>(featB, WrT, br, attR);

    // main GEMM + ReLU (WMMA, async double-buffered LDS)
    k_main<<<dim3(NNODE * BB / 128, LL / 256), 256, 0, stream>>>(featB, WmB, bm, outB);

    // edge scores + BatchNorm
    k_edge<<<NNODE * NNODE, 128, 0, stream>>>(attL, attR, gamma, beta, ebn);

    // softmax over src axis -> bf16 weights (b,d,s)
    k_soft<<<NNODE, 128, 0, stream>>>(ebn, wBuf);

    // weighted aggregation (WMMA per batch) + bias
    k_agg<<<dim3(BB, LL / 256), 256, 0, stream>>>(wBuf, outB, bias, (float*)d_out);
}
